// HiDimQuery_14216341749767
// MI455X (gfx1250) — matrix operand order repaired and checked by reading, not compiled, not run on hardware
//
#include <hip/hip_runtime.h>
#include <hip/hip_bf16.h>
#include <stdint.h>

typedef __attribute__((ext_vector_type(16))) __bf16 v16bf;
typedef __attribute__((ext_vector_type(8)))  float  v8f;

#define DIM 128
#define HID 256
#define NQ 32
#define NBITS 40
#define KNB 5
#define CHUNKS 64
#define WPB 8   // waves per block

// workspace byte offsets
#define WS_QH    0        // uint64[32]
#define WS_QSUM  256      // int[32]
#define WS_QBK   384      // float[32]
#define WS_QK    512      // float[32*128]
#define WS_BFRAG 16896    // __bf16[12*32*16]  (B fragments, bf16 WMMA layout)
#define WS_TOP   29184    // uint32[8*CHUNKS*32*5] (block-level top-5 keys)

static __device__ __forceinline__ uint32_t umin32(uint32_t a, uint32_t b){ return a<b?a:b; }
static __device__ __forceinline__ uint32_t umax32(uint32_t a, uint32_t b){ return a>b?a:b; }

// ---------------------------------------------------------------------------
// K1: sample points, query hashes, qk = Wk^T (Wq s + bq), qbk, bf16 B-frags
// ---------------------------------------------------------------------------
__global__ void k_setup(const float* __restrict__ refv, const float* __restrict__ dirs,
                        const float* __restrict__ offs, const float* __restrict__ scales,
                        const float* __restrict__ lsh,  const float* __restrict__ Wq,
                        const float* __restrict__ bq,   const float* __restrict__ Wk,
                        const float* __restrict__ bk,   char* __restrict__ ws)
{
  __shared__ float sp[NQ][DIM];
  __shared__ float qbuf[NQ][HID];
  __shared__ int   bits[NQ*NBITS];
  const int t = threadIdx.x;

  for (int e = t; e < NQ*DIM; e += 256) {
    int q = e >> 7, d = e & 127;
    int di = q >> 2, si = q & 3;
    sp[q][d] = refv[d] + dirs[di*DIM + d] * scales[di] * offs[di*4 + si];
  }
  __syncthreads();

  for (int e = t; e < NQ*NBITS; e += 256) {
    int q = e / NBITS, n = e % NBITS;
    float acc = 0.f;
    for (int d = 0; d < DIM; ++d) acc += sp[q][d] * lsh[n*DIM + d];
    bits[e] = acc > 0.f ? 1 : 0;
  }
  __syncthreads();

  if (t < NQ) {
    unsigned long long h = 0; int s = 0;
    for (int n = 0; n < NBITS; ++n) { int bb = bits[t*NBITS+n]; h |= ((unsigned long long)bb) << n; s += bb; }
    ((unsigned long long*)(ws + WS_QH))[t] = h;
    ((int*)(ws + WS_QSUM))[t] = s;
  }

  for (int e = t; e < NQ*HID; e += 256) {
    int q = e >> 8, hh = e & 255;
    float acc = bq[hh];
    for (int d = 0; d < DIM; ++d) acc += Wq[hh*DIM + d] * sp[q][d];
    qbuf[q][hh] = acc;
  }
  __syncthreads();

  for (int e = t; e < NQ*DIM; e += 256) {
    int q = e >> 7, d = e & 127;
    float acc = 0.f;
    for (int hh = 0; hh < HID; ++hh) acc += Wk[hh*DIM + d] * qbuf[q][hh];
    ((float*)(ws + WS_QK))[e] = acc;
  }
  if (t < NQ) {
    float acc = 0.f;
    for (int hh = 0; hh < HID; ++hh) acc += qbuf[t][hh] * bk[hh];
    ((float*)(ws + WS_QBK))[t] = acc;
  }

  // B fragments: B[k][n] = lsh_proj[n][k], bf16, WMMA 16x16x32 B layout.
  // lane<16: N=lane, K=kt*32+{0..7,16..23} ; lane>=16: N=lane-16, K=kt*32+{8..15,24..31}
  __bf16* bf = (__bf16*)(ws + WS_BFRAG);
  for (int e = t; e < 12*32*16; e += 256) {
    int f = e / 512;                 // nt*4 + kt
    int rem = e % 512;
    int l = rem >> 4, el = rem & 15;
    int nt = f >> 2, kt = f & 3;
    int n = nt*16 + (l & 15);
    int k = kt*32 + ((l < 16) ? 0 : 8) + ((el < 8) ? el : el + 8);
    float v = (n < NBITS) ? lsh[n*DIM + k] : 0.f;
    bf[e] = (__bf16)v;
  }
}

// ---------------------------------------------------------------------------
// K2: stream points, bf16 WMMA hash, ballot-pack bits, top-5 per (b,q)
// ---------------------------------------------------------------------------
__global__ void __launch_bounds__(256) k_main(const float* __restrict__ points,
                                              char* __restrict__ ws, int Np)
{
  const int b    = blockIdx.y;
  const int cb   = blockIdx.x;
  const int lane = threadIdx.x & 31;
  const int w    = threadIdx.x >> 5;

  const unsigned long long qh = ((const unsigned long long*)(ws + WS_QH))[lane];
  const int qsum = ((const int*)(ws + WS_QSUM))[lane];

  // resident B fragments (3 N-tiles x 4 K-steps)
  const v16bf* bsrc = (const v16bf*)(ws + WS_BFRAG);
  v16bf Bf[3][4];
#pragma unroll
  for (int nt = 0; nt < 3; ++nt)
#pragma unroll
    for (int kt = 0; kt < 4; ++kt)
      Bf[nt][kt] = bsrc[(nt*4+kt)*32 + lane];

  const int ppb = (Np + CHUNKS - 1)/CHUNKS;
  const int ppw = (ppb + WPB - 1)/WPB;
  const int blockStart = cb * ppb;
  const int blockEnd   = min(blockStart + ppb, Np);
  const int s    = blockStart + w*ppw;
  const int pEnd = min(s + ppw, blockEnd);
  const int nIter = (pEnd > s) ? (pEnd - s + 31)/32 : 0;

  const float* pbase = points + (size_t)b * Np * DIM;
  const int koff = (lane < 16) ? 0 : 8;   // A layout: lanes<16 K{0-7,16-23}, lanes>=16 K{8-15,24-31}
  const int rlow = lane & 15;

  uint32_t t0=~0u,t1=~0u,t2=~0u,t3=~0u,t4=~0u;

  for (int it = 0; it < nIter; ++it) {
    const int base = s + it*32;
    v8f acc[2][3];
#pragma unroll
    for (int mt=0; mt<2; ++mt)
#pragma unroll
      for (int nt=0; nt<3; ++nt)
#pragma unroll
        for (int e=0; e<8; ++e) acc[mt][nt][e] = 0.f;

#pragma unroll
    for (int kt = 0; kt < 4; ++kt) {
#pragma unroll
      for (int mt = 0; mt < 2; ++mt) {
        int row = base + mt*16 + rlow;
        row = row < Np ? row : Np-1;
        const float* rp = pbase + (size_t)row*DIM + kt*32 + koff;
        float4 x0 = *(const float4*)(rp);
        float4 x1 = *(const float4*)(rp + 4);
        float4 x2 = *(const float4*)(rp + 16);
        float4 x3 = *(const float4*)(rp + 20);
        v16bf a;
        a[0]=(__bf16)x0.x;  a[1]=(__bf16)x0.y;  a[2]=(__bf16)x0.z;  a[3]=(__bf16)x0.w;
        a[4]=(__bf16)x1.x;  a[5]=(__bf16)x1.y;  a[6]=(__bf16)x1.z;  a[7]=(__bf16)x1.w;
        a[8]=(__bf16)x2.x;  a[9]=(__bf16)x2.y;  a[10]=(__bf16)x2.z; a[11]=(__bf16)x2.w;
        a[12]=(__bf16)x3.x; a[13]=(__bf16)x3.y; a[14]=(__bf16)x3.z; a[15]=(__bf16)x3.w;
#pragma unroll
        for (int nt = 0; nt < 3; ++nt) {
          acc[mt][nt] = __builtin_amdgcn_wmma_f32_16x16x32_bf16(
              false, a, false, Bf[nt][kt], (short)0, acc[mt][nt], false, false);
        }
      }
    }

    // sign-ballot the C tiles: bit l<16 -> row r col l; bit l>=16 -> row r+8 col l-16
    unsigned bal[2][3][8];
#pragma unroll
    for (int mt=0; mt<2; ++mt)
#pragma unroll
      for (int nt=0; nt<3; ++nt)
#pragma unroll
        for (int r=0; r<8; ++r)
          bal[mt][nt][r] = __builtin_amdgcn_ballot_w32(acc[mt][nt][r] > 0.0f);

    // broadcast loop over the 32 points; lane = query q
#pragma unroll
    for (int p = 0; p < 32; ++p) {
      const int mt = p >> 4, r = p & 7, sh = ((p >> 3) & 1) * 16;
      const int idx = base + p;
      if (idx < pEnd) {
        unsigned long long h =
              (unsigned long long)((bal[mt][0][r] >> sh) & 0xFFFFu)
            | ((unsigned long long)((bal[mt][1][r] >> sh) & 0xFFFFu) << 16)
            | ((unsigned long long)((bal[mt][2][r] >> sh) & 0xFFu)   << 32);
        const int psum  = __popcll(h);
        const int cross = __popcll(h & qh);
        const int dist  = qsum + psum - 2*cross;        // Hamming, 0..40
        const uint32_t key = ((uint32_t)dist << 17) | (uint32_t)idx;
        if (__builtin_amdgcn_ballot_w32(key < t4)) {
          uint32_t c = key, m;
          m = umin32(t0,c); c = umax32(t0,c); t0 = m;
          m = umin32(t1,c); c = umax32(t1,c); t1 = m;
          m = umin32(t2,c); c = umax32(t2,c); t2 = m;
          m = umin32(t3,c); c = umax32(t3,c); t3 = m;
          t4 = umin32(t4,c);
        }
      }
    }
  }

  __shared__ uint32_t lt[WPB][NQ][KNB];
  lt[w][lane][0]=t0; lt[w][lane][1]=t1; lt[w][lane][2]=t2; lt[w][lane][3]=t3; lt[w][lane][4]=t4;
  __syncthreads();
  if (threadIdx.x < NQ) {
    const int q = threadIdx.x;
    uint32_t r0=~0u,r1=~0u,r2=~0u,r3=~0u,r4=~0u;
    for (int ww = 0; ww < WPB; ++ww)
      for (int i = 0; i < KNB; ++i) {
        uint32_t c = lt[ww][q][i], m;
        m = umin32(r0,c); c = umax32(r0,c); r0 = m;
        m = umin32(r1,c); c = umax32(r1,c); r1 = m;
        m = umin32(r2,c); c = umax32(r2,c); r2 = m;
        m = umin32(r3,c); c = umax32(r3,c); r3 = m;
        r4 = umin32(r4,c);
      }
    uint32_t* wt = (uint32_t*)(ws + WS_TOP) + (((size_t)b*CHUNKS + cb)*NQ + q)*KNB;
    wt[0]=r0; wt[1]=r1; wt[2]=r2; wt[3]=r3; wt[4]=r4;
  }
}

// ---------------------------------------------------------------------------
// K3: merge block partials, attention tail (one block per batch)
// ---------------------------------------------------------------------------
__global__ void k_final(const float* __restrict__ points, const float* __restrict__ Wv,
                        const float* __restrict__ bv,     const float* __restrict__ Wo,
                        const float* __restrict__ bo,     const char* __restrict__ ws,
                        float* __restrict__ out, int Np)
{
  const int b = blockIdx.x;
  const int t = threadIdx.x;
  __shared__ int   nidx[NQ][KNB];
  __shared__ float logit[NQ][KNB];
  __shared__ float attw[NQ][KNB];
  __shared__ float mwnb[DIM];
  __shared__ float tvec[HID];

  if (t < NQ) {
    uint32_t r0=~0u,r1=~0u,r2=~0u,r3=~0u,r4=~0u;
    for (int cb = 0; cb < CHUNKS; ++cb) {
      const uint32_t* wt = (const uint32_t*)(ws + WS_TOP) + (((size_t)b*CHUNKS + cb)*NQ + t)*KNB;
      for (int i = 0; i < KNB; ++i) {
        uint32_t c = wt[i], m;
        m = umin32(r0,c); c = umax32(r0,c); r0 = m;
        m = umin32(r1,c); c = umax32(r1,c); r1 = m;
        m = umin32(r2,c); c = umax32(r2,c); r2 = m;
        m = umin32(r3,c); c = umax32(r3,c); r3 = m;
        r4 = umin32(r4,c);
      }
    }
    nidx[t][0] = (int)(r0 & 0x1FFFFu);
    nidx[t][1] = (int)(r1 & 0x1FFFFu);
    nidx[t][2] = (int)(r2 & 0x1FFFFu);
    nidx[t][3] = (int)(r3 & 0x1FFFFu);
    nidx[t][4] = (int)(r4 & 0x1FFFFu);
  }
  __syncthreads();

  const float* qk  = (const float*)(ws + WS_QK);
  const float* qbk = (const float*)(ws + WS_QBK);
  const float* pb  = points + (size_t)b * Np * DIM;

  if (t < NQ*KNB) {
    int q = t / KNB, k = t % KNB;
    const float* nb = pb + (size_t)nidx[q][k]*DIM;
    float acc = 0.f;
    for (int d = 0; d < DIM; ++d) acc += qk[q*DIM + d]*nb[d];
    logit[q][k] = (acc + qbk[q]) * (1.0f/16.0f);   // sqrt(256)=16
  }
  __syncthreads();

  if (t < NQ) {
    float mx = logit[t][0];
    for (int k=1;k<KNB;++k) mx = fmaxf(mx, logit[t][k]);
    float e[KNB]; float sum=0.f;
    for (int k=0;k<KNB;++k){ e[k]=expf(logit[t][k]-mx); sum+=e[k]; }
    for (int k=0;k<KNB;++k){
      float wv = e[k]/sum;
      attw[t][k] = wv;
      out[1024 + ((size_t)b*NQ + t)*KNB + k] = wv;
    }
  }
  __syncthreads();

  if (t < DIM) {
    float acc = 0.f;
    for (int q=0;q<NQ;++q)
      for (int k=0;k<KNB;++k)
        acc += attw[q][k] * pb[(size_t)nidx[q][k]*DIM + t];
    mwnb[t] = acc * (1.0f/NQ);
  }
  __syncthreads();

  {
    float acc = bv[t];
    for (int d=0; d<DIM; ++d) acc += Wv[t*DIM + d]*mwnb[d];
    tvec[t] = acc;
  }
  __syncthreads();

  if (t < DIM) {
    float acc = bo[t];
    for (int hh=0; hh<HID; ++hh) acc += Wo[t*HID + hh]*tvec[hh];
    out[(size_t)b*DIM + t] = acc;
  }
}

// ---------------------------------------------------------------------------
extern "C" void kernel_launch(void* const* d_in, const int* in_sizes, int n_in,
                              void* d_out, int out_size, void* d_ws, size_t ws_size,
                              hipStream_t stream)
{
  const float* points = (const float*)d_in[0];
  const float* refv   = (const float*)d_in[1];
  const float* dirs   = (const float*)d_in[2];
  const float* offs   = (const float*)d_in[3];
  const float* scales = (const float*)d_in[4];
  const float* lsh    = (const float*)d_in[5];
  const float* Wq     = (const float*)d_in[6];
  const float* bq     = (const float*)d_in[7];
  const float* Wk     = (const float*)d_in[8];
  const float* bk     = (const float*)d_in[9];
  const float* Wv     = (const float*)d_in[10];
  const float* bv     = (const float*)d_in[11];
  const float* Wo     = (const float*)d_in[12];
  const float* bo     = (const float*)d_in[13];

  const int B  = 8;
  const int Np = in_sizes[0] / (B * DIM);
  char*  ws  = (char*)d_ws;
  float* out = (float*)d_out;

  k_setup<<<1, 256, 0, stream>>>(refv, dirs, offs, scales, lsh, Wq, bq, Wk, bk, ws);
  k_main <<<dim3(CHUNKS, B), 256, 0, stream>>>(points, ws, Np);
  k_final<<<B, 256, 0, stream>>>(points, Wv, bv, Wo, bo, ws, out, Np);
}
